// WordInQuestionW_38551626449363
// MI455X (gfx1250) — compile-verified, matrix operand order checked by reading, not run on hardware
//
#include <hip/hip_runtime.h>

#define NB 32
#define NC 4096
#define NQ 64
#define ND 256

// WMMA vector types (gfx1250, wave32)
typedef __attribute__((ext_vector_type(16))) __bf16 v16bf;
typedef __attribute__((ext_vector_type(8)))  float  v8f;

// Padded LDS strides
#define QSTRIDE 264   // 256 bf16 + 8 pad  -> lane bank stride 4, row = 528 B (8B aligned)
#define SSTRIDE 68    // 64 f32 + 4 pad    -> row base stays 16B aligned (68*4=272)

// One-instruction bf16x2 pack (truncate): bytes {lo[2],lo[3],hi[2],hi[3]} via v_perm_b32
__device__ __forceinline__ unsigned pack2bf(float lo, float hi) {
    return __builtin_amdgcn_perm(__float_as_uint(hi), __float_as_uint(lo), 0x07060302u);
}

__global__ __launch_bounds__(256)
void wiq_softmax_kernel(const float* __restrict__ question,
                        const float* __restrict__ context,
                        const float* __restrict__ weight,
                        float* __restrict__ out) {
    // Shared memory: phase 1/2 uses it as bf16 question*w [64][QSTRIDE];
    // phase 3 (after barrier) reuses it as f32 sim tile [128][SSTRIDE].
    __shared__ __align__(16) unsigned char smem[128 * SSTRIDE * 4]; // 34816 B >= 64*QSTRIDE*2
    unsigned short* qlds = (unsigned short*)smem;
    float*          slds = (float*)smem;

    const int tid  = threadIdx.x;
    const int lane = tid & 31;
    const int wid  = tid >> 5;          // 8 waves
    const int half = lane >> 4;         // 0: lanes 0-15, 1: lanes 16-31
    const int lrow = lane & 15;

    const int b      = blockIdx.x >> 5; // 32 batches
    const int c_base = (blockIdx.x & 31) * 128;

    // ---- Stage 1: qlds[q][d] = bf16( question[b][q][d] * w[d] ), vectorized ----
    {
        const float4* qb4 = (const float4*)(question + (size_t)b * NQ * ND);
        const float4  w4  = ((const float4*)weight)[tid & 63];
        const int d4 = (tid & 63) * 4;   // column group (in floats)
        const int q0 = tid >> 6;         // starting row
        #pragma unroll 4
        for (int i = 0; i < 16; ++i) {
            const int q = q0 + 4 * i;
            float4 v = qb4[q * (ND / 4) + (tid & 63)];   // coalesced b128 stream
            unsigned lo = pack2bf(v.x * w4.x, v.y * w4.y);
            unsigned hi = pack2bf(v.z * w4.z, v.w * w4.w);
            uint2 pk; pk.x = lo; pk.y = hi;
            *(uint2*)(qlds + q * QSTRIDE + d4) = pk;     // ds_store_b64, 8B aligned
        }
    }
    __syncthreads();

    // ---- Stage 2: each wave computes a 16x64 sim tile via WMMA bf16 ----
    const float* arow =
        context + ((size_t)b * NC + (size_t)(c_base + wid * 16 + lrow)) * ND;

    v8f acc[4];
    #pragma unroll
    for (int nt = 0; nt < 4; ++nt) acc[nt] = (v8f){0.f,0.f,0.f,0.f,0.f,0.f,0.f,0.f};

    #pragma unroll
    for (int ks = 0; ks < 8; ++ks) {
        const int kb = ks * 32 + half * 8;
        if (ks < 7) __builtin_prefetch(arow + kb + 32, 0, 1); // global_prefetch_b8

        // A operand: 16x32 bf16 per ISA layout (lanes 0-15 K0-7/16-23, 16-31 K8-15/24-31)
        const float4* p0 = (const float4*)(arow + kb);
        const float4* p1 = (const float4*)(arow + kb + 16);
        float4 f0 = p0[0], f1 = p0[1], f2 = p1[0], f3 = p1[1];
        union { v16bf v; unsigned u[8]; } A;
        A.u[0] = pack2bf(f0.x, f0.y); A.u[1] = pack2bf(f0.z, f0.w);
        A.u[2] = pack2bf(f1.x, f1.y); A.u[3] = pack2bf(f1.z, f1.w);
        A.u[4] = pack2bf(f2.x, f2.y); A.u[5] = pack2bf(f2.z, f2.w);
        A.u[6] = pack2bf(f3.x, f3.y); A.u[7] = pack2bf(f3.z, f3.w);

        #pragma unroll
        for (int nt = 0; nt < 4; ++nt) {
            // B operand = qw^T (K x N): lane holds column n = nt*16+lrow, same K split as A
            const unsigned short* brow =
                qlds + (nt * 16 + lrow) * QSTRIDE + ks * 32 + half * 8;
            union { v16bf v; unsigned u[8]; } Bv;
            Bv.u[0] = *(const unsigned*)(brow + 0);
            Bv.u[1] = *(const unsigned*)(brow + 2);
            Bv.u[2] = *(const unsigned*)(brow + 4);
            Bv.u[3] = *(const unsigned*)(brow + 6);
            Bv.u[4] = *(const unsigned*)(brow + 16);
            Bv.u[5] = *(const unsigned*)(brow + 18);
            Bv.u[6] = *(const unsigned*)(brow + 20);
            Bv.u[7] = *(const unsigned*)(brow + 22);

            acc[nt] = __builtin_amdgcn_wmma_f32_16x16x32_bf16(
                false, A.v, false, Bv.v, (short)0, acc[nt], false, false);
        }
    }

    __syncthreads(); // all qlds reads done -> safe to reuse smem as sim tile

    // ---- Stage 3: spill accumulators (C/D layout: row = r + 8*half, col = lrow) ----
    {
        const int m0 = wid * 16 + half * 8;
        #pragma unroll
        for (int nt = 0; nt < 4; ++nt) {
            #pragma unroll
            for (int r = 0; r < 8; ++r) {
                slds[(m0 + r) * SSTRIDE + nt * 16 + lrow] = acc[nt][r];
            }
        }
    }
    __syncthreads();

    // ---- Stage 4: per-row softmax over q (64) then sum -> one scalar per c-row ----
    if (tid < 128) {
        const float4* rp = (const float4*)(slds + tid * SSTRIDE); // 272B-aligned
        float mx = -3.402823466e+38f;
        #pragma unroll
        for (int i = 0; i < 16; ++i) {
            float4 v = rp[i];
            mx = fmaxf(mx, fmaxf(fmaxf(v.x, v.y), fmaxf(v.z, v.w)));
        }
        float s = 0.f;
        #pragma unroll
        for (int i = 0; i < 16; ++i) {
            float4 v = rp[i];
            s += __expf(v.x - mx) + __expf(v.y - mx) +
                 __expf(v.z - mx) + __expf(v.w - mx);
        }
        const float inv = 1.0f / s;
        float r = 0.f;
        #pragma unroll
        for (int i = 0; i < 16; ++i) {
            float4 v = rp[i];
            r += __expf(v.x - mx) * inv + __expf(v.y - mx) * inv +
                 __expf(v.z - mx) * inv + __expf(v.w - mx) * inv;
        }
        out[(size_t)b * NC + c_base + tid] = r;
    }
}

extern "C" void kernel_launch(void* const* d_in, const int* in_sizes, int n_in,
                              void* d_out, int out_size, void* d_ws, size_t ws_size,
                              hipStream_t stream) {
    (void)in_sizes; (void)n_in; (void)out_size; (void)d_ws; (void)ws_size;
    const float* question = (const float*)d_in[0];
    const float* context  = (const float*)d_in[1];
    const float* weight   = (const float*)d_in[2];
    float* out = (float*)d_out;

    dim3 grid(NB * (NC / 128)); // 1024 workgroups
    dim3 block(256);            // 8 wave32s
    wiq_softmax_kernel<<<grid, block, 0, stream>>>(question, context, weight, out);
}
